// RangeViewSAModuleMSG_52682068853179
// MI455X (gfx1250) — matrix operand order repaired and checked by reading, not compiled, not run on hardware
//
#include <hip/hip_runtime.h>
#include <hip/hip_fp16.h>

typedef __attribute__((ext_vector_type(16))) _Float16 v16h;
typedef __attribute__((ext_vector_type(8)))  float    v8f;

#define BN_EPS 1e-5f

// sigmoid denominator is in [1, 2) for x>=0 and (1, inf) for x<0 -- never
// denormal/zero, so the raw v_rcp_f32 is safe (avoids v_div_scale/fixup).
__device__ __forceinline__ float silu_f(float x) {
    return x * __builtin_amdgcn_rcpf(1.0f + __expf(-x));
}

// Lane-XOR exchange via ds_swizzle_b32 (group-of-32 mode, immediate pattern:
// offset = xor_mask<<10 | or_mask<<5 | and_mask; and_mask=0x1f keeps all lane
// bits). One DS op, no per-call address-VGPR setup like __shfl_xor emits.
template <int XOR_MASK>
__device__ __forceinline__ float swz_xor(float v) {
    constexpr int pat = (XOR_MASK << 10) | 0x1f;
    return __int_as_float(__builtin_amdgcn_ds_swizzle(__float_as_int(v), pat));
}

// One wave (32 lanes) handles one query center. Per 16-sample tile:
//   B0 (32x16 f16, K=19 padded)  <- gathered [rel_xyz(3) | features(16)]
//   D0 = WMMA(A0=W0, B0)         -> BN + SiLU (per-element on accumulator)
//   B1 built from D0 via one ds_swizzle XOR-16 across lane halves
//   D1 = WMMA(A1=W1 block, B1)   -> BN + SiLU -> running max over samples
// Final: max-reduce across the 16 columns (XOR swizzles within 16-lane halves).
template <int NS, int COUT1>
__global__ __launch_bounds__(256)
void sa_branch_wmma(const float* __restrict__ xyz,
                    const float* __restrict__ feat,
                    const float* __restrict__ qxyz,
                    const int*   __restrict__ idx,
                    const unsigned char* __restrict__ empty,
                    const float* __restrict__ W0,
                    const float* __restrict__ g0,  const float* __restrict__ be0,
                    const float* __restrict__ mu0, const float* __restrict__ va0,
                    const float* __restrict__ W1,
                    const float* __restrict__ g1,  const float* __restrict__ be1,
                    const float* __restrict__ mu1, const float* __restrict__ va1,
                    float* __restrict__ out, int colBase, int M)
{
    constexpr int KB = COUT1 / 16;
    const int  lane = threadIdx.x & 31;
    const int  row  = lane & 15;
    const bool hi   = lane >= 16;
    const int  q    = blockIdx.x * 8 + (threadIdx.x >> 5);
    if (q >= M) return;            // wave-uniform: EXEC stays all-1s for WMMA

    // ---- A0 fragment: layer-0 weights W0 (16x19), K padded to 32 ----
    v16h a0 = {};
#pragma unroll
    for (int j = 0; j < 16; ++j) {
        const int k = j + (hi ? 16 : 0);
        if (k < 19) a0[j] = (_Float16)W0[row * 19 + k];
    }

    // ---- A1 fragments: layer-1 weights W1 (COUT1 x 16), K padded to 32 ----
    v16h a1[KB] = {};
#pragma unroll
    for (int kb = 0; kb < KB; ++kb)
#pragma unroll
        for (int j = 0; j < 16; ++j)
            if (!hi) a1[kb][j] = (_Float16)W1[(kb * 16 + row) * 16 + j];

    // ---- folded BN constants for this lane's accumulator channels ----
    // var + eps >= 1e-5 is never denormal -> raw v_rsq_f32 is safe.
    float inv0[8], bet0[8];
#pragma unroll
    for (int r = 0; r < 8; ++r) {
        const int ch = (hi ? 8 : 0) + r;
        const float iv = g0[ch] * __builtin_amdgcn_rsqf(va0[ch] + BN_EPS);
        inv0[r] = iv;
        bet0[r] = be0[ch] - mu0[ch] * iv;
    }
    float inv1[KB][8], bet1[KB][8];
#pragma unroll
    for (int kb = 0; kb < KB; ++kb)
#pragma unroll
        for (int r = 0; r < 8; ++r) {
            const int ch = kb * 16 + (hi ? 8 : 0) + r;
            const float iv = g1[ch] * __builtin_amdgcn_rsqf(va1[ch] + BN_EPS);
            inv1[kb][r] = iv;
            bet1[kb][r] = be1[ch] - mu1[ch] * iv;
        }

    const float qx = qxyz[q * 3 + 0];
    const float qy = qxyz[q * 3 + 1];
    const float qz = qxyz[q * 3 + 2];
    const bool isEmpty = empty[q] != 0;   // wave-uniform

    float pmax[KB][8];
#pragma unroll
    for (int kb = 0; kb < KB; ++kb)
#pragma unroll
        for (int r = 0; r < 8; ++r) pmax[kb][r] = -3.4e38f;

#pragma unroll
    for (int sb = 0; sb < NS; sb += 16) {
        // ---- gather B0: lanes n and n+16 share column n (sample sb+n) ----
        v16h b0 = {};
        if (!isEmpty) {
            const int p = idx[q * NS + sb + row];
            const float4* f4 = reinterpret_cast<const float4*>(feat + p * 16);
            const float4 fA = f4[0];
            const float4 fB = f4[1];
            const float4 fC = f4[2];
            const float4 fD = f4[3];
            if (!hi) {   // K = 0..15 : rel xyz + features 0..12
                b0[0]  = (_Float16)(xyz[p * 3 + 0] - qx);
                b0[1]  = (_Float16)(xyz[p * 3 + 1] - qy);
                b0[2]  = (_Float16)(xyz[p * 3 + 2] - qz);
                b0[3]  = (_Float16)fA.x;  b0[4]  = (_Float16)fA.y;
                b0[5]  = (_Float16)fA.z;  b0[6]  = (_Float16)fA.w;
                b0[7]  = (_Float16)fB.x;  b0[8]  = (_Float16)fB.y;
                b0[9]  = (_Float16)fB.z;  b0[10] = (_Float16)fB.w;
                b0[11] = (_Float16)fC.x;  b0[12] = (_Float16)fC.y;
                b0[13] = (_Float16)fC.z;  b0[14] = (_Float16)fC.w;
                b0[15] = (_Float16)fD.x;
            } else {     // K = 16..18 : features 13..15 (19..31 zero pad)
                b0[0] = (_Float16)fD.y;
                b0[1] = (_Float16)fD.z;
                b0[2] = (_Float16)fD.w;
            }
        }

        // ---- layer 0: 16x16x32 f16 WMMA, f32 accumulate ----
        v8f c0 = {};
        c0 = __builtin_amdgcn_wmma_f32_16x16x32_f16(false, a0, false, b0,
                                                    (short)0, c0, false, false);
        float act[8];
#pragma unroll
        for (int r = 0; r < 8; ++r)
            act[r] = silu_f(c0[r] * inv0[r] + bet0[r]);

        // ---- transpose-free repack D0 -> B1 (one swizzle across halves) ----
        v16h b1 = {};
#pragma unroll
        for (int r = 0; r < 8; ++r) {
            const float partner = swz_xor<16>(act[r]);  // channels 8..15 of my column
            if (!hi) {
                b1[r]     = (_Float16)act[r];
                b1[8 + r] = (_Float16)partner;
            }
        }

        // ---- layer 1: one WMMA per 16-row output block ----
#pragma unroll
        for (int kb = 0; kb < KB; ++kb) {
            v8f c1 = {};
            c1 = __builtin_amdgcn_wmma_f32_16x16x32_f16(false, a1[kb], false, b1,
                                                        (short)0, c1, false, false);
#pragma unroll
            for (int r = 0; r < 8; ++r) {
                const float y = silu_f(c1[r] * inv1[kb][r] + bet1[kb][r]);
                pmax[kb][r] = fmaxf(pmax[kb][r], y);
            }
        }
    }

    // ---- max over 16 sample-columns: stays within each 16-lane half ----
#pragma unroll
    for (int kb = 0; kb < KB; ++kb)
#pragma unroll
        for (int r = 0; r < 8; ++r) {
            float v = pmax[kb][r];
            v = fmaxf(v, swz_xor<8>(v));
            v = fmaxf(v, swz_xor<4>(v));
            v = fmaxf(v, swz_xor<2>(v));
            v = fmaxf(v, swz_xor<1>(v));
            pmax[kb][r] = v;
        }

    // lanes 0 and 16 hold the 8-channel results for their half
    if (row == 0) {
#pragma unroll
        for (int kb = 0; kb < KB; ++kb)
#pragma unroll
            for (int r = 0; r < 8; ++r)
                out[q * 48 + colBase + kb * 16 + (hi ? 8 : 0) + r] = pmax[kb][r];
    }
}

extern "C" void kernel_launch(void* const* d_in, const int* in_sizes, int n_in,
                              void* d_out, int out_size, void* d_ws, size_t ws_size,
                              hipStream_t stream) {
    (void)n_in; (void)out_size; (void)d_ws; (void)ws_size;
    // setup_inputs() order:
    // 0 xyz(N,3) 1 features(N,16) 2 query_xyz(M,3) 3 idx0(M,16) 4 idx1(M,32)
    // 5 empty0(M) 6 empty1(M)
    // 7..11  mlp0 layer0 (W 16x19, g,b,mu,var 16)
    // 12..16 mlp0 layer1 (W 16x16, g,b,mu,var 16)
    // 17..21 mlp1 layer0 (W 16x19, g,b,mu,var 16)
    // 22..26 mlp1 layer1 (W 32x16, g,b,mu,var 32)
    const float* xyz    = (const float*)d_in[0];
    const float* feat   = (const float*)d_in[1];
    const float* qxyz   = (const float*)d_in[2];
    const int*   idx0   = (const int*)d_in[3];
    const int*   idx1   = (const int*)d_in[4];
    const unsigned char* empty0 = (const unsigned char*)d_in[5];
    const unsigned char* empty1 = (const unsigned char*)d_in[6];
    const int M = in_sizes[2] / 3;

    float* out = (float*)d_out;
    dim3 block(256);                 // 8 wave32 = 8 queries per block
    dim3 grid((M + 7) / 8);

    sa_branch_wmma<16, 16><<<grid, block, 0, stream>>>(
        xyz, feat, qxyz, idx0, empty0,
        (const float*)d_in[7],  (const float*)d_in[8],  (const float*)d_in[9],
        (const float*)d_in[10], (const float*)d_in[11],
        (const float*)d_in[12], (const float*)d_in[13], (const float*)d_in[14],
        (const float*)d_in[15], (const float*)d_in[16],
        out, /*colBase=*/0, M);

    sa_branch_wmma<32, 32><<<grid, block, 0, stream>>>(
        xyz, feat, qxyz, idx1, empty1,
        (const float*)d_in[17], (const float*)d_in[18], (const float*)d_in[19],
        (const float*)d_in[20], (const float*)d_in[21],
        (const float*)d_in[22], (const float*)d_in[23], (const float*)d_in[24],
        (const float*)d_in[25], (const float*)d_in[26],
        out, /*colBase=*/16, M);
}
